// GIBGIN_81621558493401
// MI455X (gfx1250) — matrix operand order repaired and checked by reading, not compiled, not run on hardware
//
#include <hip/hip_runtime.h>
#include <hip/hip_bf16.h>

// ---------------------------------------------------------------------------
// GIN forward on gfx1250: WMMA f32_16x16x32_f16 for the per-layer MLP GEMMs.
// Weights pre-packed into WMMA B-fragment layout -> b128 loads, no VALU packing.
// ---------------------------------------------------------------------------

#define NN 100000
#define EE 400000
#define GG 2048
#define HN 256
#define H2 512
#define NL 5

typedef __attribute__((ext_vector_type(16))) _Float16 v16h;
typedef __attribute__((ext_vector_type(8)))  float    v8f;

union AFrag { v16h v; unsigned u[8]; uint4 q[2]; };
union H2U   { _Float16 h[2]; unsigned u; };

// A-fragment (16x32 f16) from LDS, contiguous K to match packed B: lanes 0-15 row M=lane,
// K 0..15 in v0..7 (2/vgpr); lanes 16-31 K 16..31.
__device__ inline v16h load_a_frag_lds(const _Float16* base, int stride,
                                       int row, int half, int k) {
  AFrag f;
  #pragma unroll
  for (int v = 0; v < 4; ++v) {
    int kk = k + half * 16 + 2 * v;
    f.u[v] = *(const unsigned*)(base + row * stride + kk);
  }
  #pragma unroll
  for (int v = 0; v < 4; ++v) {
    int kk = k + half * 16 + 8 + 2 * v;
    f.u[4 + v] = *(const unsigned*)(base + row * stride + kk);
  }
  return f.v;
}

// B-fragment from pre-packed weights: 32 contiguous bytes per lane.
__device__ inline v16h load_b_frag_packed(const unsigned* __restrict__ wp,
                                          long frag, int lane) {
  const uint4* p = (const uint4*)(wp + (frag * 32 + lane) * 8);
  AFrag f;
  f.q[0] = p[0];
  f.q[1] = p[1];
  return f.v;
}

// ---------------------------------------------------------------------------
// Prep: pack f32 weights into WMMA B-fragment-major f16 layout; fold b1 +
// batchnorm into per-column scale/shift.
// Packed layout: uint index = ((l*nTiles + nt)*kSteps + ks)*256 + lane*8 + v
//   half = lane>>4, col = lane&15, kk = ks*32 + half*16 + 2v, colg = nt*16+col
//   value = pack(f16(W[l][kk][colg]), f16(W[l][kk+1][colg]))
__global__ void gin_prep_k(const float* __restrict__ W1, const float* __restrict__ W2,
                           const float* __restrict__ b1, const float* __restrict__ gamma,
                           const float* __restrict__ beta, const float* __restrict__ mean,
                           const float* __restrict__ var,
                           unsigned* __restrict__ w1p, unsigned* __restrict__ w2p,
                           float* __restrict__ bnscale, float* __restrict__ bnshift) {
  int idx = blockIdx.x * 256 + threadIdx.x;
  const int nP = NL * 32 * 8 * 32 * 8;   // 327680 packed dwords per matrix set
  if (idx < nP) {
    int l    = idx >> 16;
    int rem  = idx & 65535;
    int lane = (rem >> 3) & 31;
    int v    = rem & 7;
    int half = lane >> 4, col = lane & 15;
    {
      // W1: [256, 512], nTiles=32, kSteps=8
      int nt = rem >> 11;
      int ks = (rem >> 8) & 7;
      int kk = ks * 32 + half * 16 + 2 * v;
      int colg = nt * 16 + col;
      H2U t;
      t.h[0] = (_Float16)W1[((long)(l * HN + kk)) * H2 + colg];
      t.h[1] = (_Float16)W1[((long)(l * HN + kk + 1)) * H2 + colg];
      w1p[idx] = t.u;
    }
    {
      // W2: [512, 256], nTiles=16, kSteps=16
      int nt = rem >> 12;
      int ks = (rem >> 8) & 15;
      int kk = ks * 32 + half * 16 + 2 * v;
      int colg = nt * 16 + col;
      H2U t;
      t.h[0] = (_Float16)W2[((long)(l * H2 + kk)) * HN + colg];
      t.h[1] = (_Float16)W2[((long)(l * H2 + kk + 1)) * HN + colg];
      w2p[idx] = t.u;
    }
  }
  if (idx < NL * H2) {
    float sc = gamma[idx] * rsqrtf(var[idx] + 1e-5f);
    bnscale[idx] = sc;
    bnshift[idx] = beta[idx] + (b1[idx] - mean[idx]) * sc;
  }
}

__global__ void gin_zero_k(float* __restrict__ p, long n) {
  long i = (long)blockIdx.x * blockDim.x + threadIdx.x;
  long stride = (long)gridDim.x * blockDim.x;
  for (; i < n; i += stride) p[i] = 0.0f;
}

// AtomEncoder: h[n,c] = sum_f atom_emb[f, x[n,f], c]
__global__ __launch_bounds__(256) void gin_atom_enc(const int* __restrict__ x,
                                                    const float* __restrict__ atom_emb,
                                                    float* __restrict__ h) {
  int n = blockIdx.x, c = threadIdx.x;
  float s = 0.0f;
  #pragma unroll
  for (int f = 0; f < 9; ++f) {
    int v = x[n * 9 + f];
    s += atom_emb[((long)(f * 128 + v)) * HN + c];
  }
  h[(long)n * HN + c] = s;
}

// Edge messages: msg = relu(h[src] + bond_emb_sum) scattered into agg[dst].
__global__ __launch_bounds__(256) void gin_edge_msg(const int* __restrict__ ei,
                                                    const int* __restrict__ ea,
                                                    const float* __restrict__ bond_l,
                                                    const float* __restrict__ h,
                                                    float* __restrict__ agg) {
  int e = blockIdx.x, c = threadIdx.x;
  int src = ei[e];
  int dst = ei[EE + e];
  float s = h[(long)src * HN + c];
  #pragma unroll
  for (int f = 0; f < 3; ++f) {
    int v = ea[e * 3 + f];
    s += bond_l[((long)(f * 16 + v)) * HN + c];
  }
  s = fmaxf(s, 0.0f);
  atomicAdd(&agg[(long)dst * HN + c], s);
}

// ---------------------------------------------------------------------------
// Fused GIN MLP per 16-row strip:
//   z = (1+eps)*h + agg  (f16 in LDS)
//   mid = relu(BN(z @ W1 + b1))  (f16 in LDS, 16x512)
//   h'  = mid @ W2 + b2  (f32 to global)
// 8 waves/block; GEMM1: 4 x 16x16 tiles/wave; GEMM2: 2 tiles/wave.
__global__ __launch_bounds__(256) void gin_mlp_fused(
    const float* __restrict__ hin, const float* __restrict__ agg,
    const float* __restrict__ epsv, int layer,
    const unsigned* __restrict__ w1p, const unsigned* __restrict__ w2p,
    const float* __restrict__ bnscale, const float* __restrict__ bnshift,
    const float* __restrict__ b2, float* __restrict__ hout) {
  __shared__ _Float16 sIn[16 * HN];    // 8 KB
  __shared__ _Float16 sMid[16 * H2];   // 16 KB
  int tid = threadIdx.x;
  int wave = tid >> 5, lane = tid & 31;
  int half = lane >> 4, lc = lane & 15;
  long mBase = (long)blockIdx.x * 16;
  float epsl = 1.0f + epsv[layer];

  // Stage A strip (rows are contiguous): float4 reads, packed dword LDS writes.
  {
    const float4* hp = (const float4*)(hin + mBase * HN);
    const float4* ap = (const float4*)(agg + mBase * HN);
    unsigned* sp = (unsigned*)sIn;
    for (int i = tid; i < (16 * HN) / 4; i += 256) {
      float4 hv = hp[i], av = ap[i];
      H2U p0, p1;
      p0.h[0] = (_Float16)(epsl * hv.x + av.x);
      p0.h[1] = (_Float16)(epsl * hv.y + av.y);
      p1.h[0] = (_Float16)(epsl * hv.z + av.z);
      p1.h[1] = (_Float16)(epsl * hv.w + av.w);
      sp[i * 2 + 0] = p0.u;
      sp[i * 2 + 1] = p1.u;
    }
  }
  __syncthreads();

  // ---- GEMM1: [16,256] @ [256,512]  (nTiles=32, kSteps=8) ----
  v8f acc[4];
  #pragma unroll
  for (int j = 0; j < 4; ++j)
    #pragma unroll
    for (int i = 0; i < 8; ++i) acc[j][i] = 0.0f;

  for (int ks = 0; ks < 8; ++ks) {
    v16h a = load_a_frag_lds(sIn, HN, lc, half, ks * 32);
    #pragma unroll
    for (int j = 0; j < 4; ++j) {
      int nt = wave * 4 + j;
      long frag = ((long)(layer * 32 + nt)) * 8 + ks;
      v16h b = load_b_frag_packed(w1p, frag, lane);
      acc[j] = __builtin_amdgcn_wmma_f32_16x16x32_f16(false, a, false, b,
                                                      (short)0, acc[j], false, false);
    }
  }
  #pragma unroll
  for (int j = 0; j < 4; ++j) {
    int colg = (wave * 4 + j) * 16 + lc;
    float sc = bnscale[layer * H2 + colg];
    float sh = bnshift[layer * H2 + colg];
    #pragma unroll
    for (int v = 0; v < 8; ++v) {
      int r = half * 8 + v;
      float f = acc[j][v] * sc + sh;
      sMid[r * H2 + colg] = (_Float16)fmaxf(f, 0.0f);
    }
  }
  __syncthreads();

  // ---- GEMM2: [16,512] @ [512,256]  (nTiles=16, kSteps=16) ----
  v8f acc2[2];
  #pragma unroll
  for (int j = 0; j < 2; ++j)
    #pragma unroll
    for (int i = 0; i < 8; ++i) acc2[j][i] = 0.0f;

  for (int ks = 0; ks < 16; ++ks) {
    v16h a = load_a_frag_lds(sMid, H2, lc, half, ks * 32);
    #pragma unroll
    for (int j = 0; j < 2; ++j) {
      int nt = wave * 2 + j;
      long frag = ((long)(layer * 16 + nt)) * 16 + ks;
      v16h b = load_b_frag_packed(w2p, frag, lane);
      acc2[j] = __builtin_amdgcn_wmma_f32_16x16x32_f16(false, a, false, b,
                                                       (short)0, acc2[j], false, false);
    }
  }
  #pragma unroll
  for (int j = 0; j < 2; ++j) {
    int colg = (wave * 2 + j) * 16 + lc;
    float bb = b2[layer * HN + colg];
    #pragma unroll
    for (int v = 0; v < 8; ++v) {
      int r = half * 8 + v;
      hout[(mBase + r) * HN + colg] = acc2[j][v] + bb;
    }
  }
}

// ---------------------------------------------------------------------------
// assignment = softmax(tanh(h@c1W + c1b) @ c2W + c2b); mask = a0 > 0.5
__global__ __launch_bounds__(256) void gin_assign_k(
    const float* __restrict__ h, const float* __restrict__ c1W,
    const float* __restrict__ c1b, const float* __restrict__ c2W,
    const float* __restrict__ c2b, float* __restrict__ assignp,
    float* __restrict__ mask) {
  __shared__ float hs[256];
  __shared__ float r0[256], r1[256];
  int n = blockIdx.x, c = threadIdx.x;
  hs[c] = h[(long)n * HN + c];
  __syncthreads();
  float acc = c1b[c];
  for (int k = 0; k < HN; ++k) acc += hs[k] * c1W[k * HN + c];
  float t = tanhf(acc);
  r0[c] = t * c2W[c * 2 + 0];
  r1[c] = t * c2W[c * 2 + 1];
  __syncthreads();
  for (int s = 128; s > 0; s >>= 1) {
    if (c < s) { r0[c] += r0[c + s]; r1[c] += r1[c + s]; }
    __syncthreads();
  }
  if (c == 0) {
    float z0 = r0[0] + c2b[0], z1 = r1[0] + c2b[1];
    float m = fmaxf(z0, z1);
    float e0 = expf(z0 - m), e1 = expf(z1 - m);
    float a0 = e0 / (e0 + e1);
    assignp[(long)n * 2 + 0] = a0;
    assignp[(long)n * 2 + 1] = 1.0f - a0;
    mask[n] = (a0 > 0.5f) ? 1.0f : 0.0f;
  }
}

__global__ __launch_bounds__(256) void gin_pool_k(
    const float* __restrict__ h, const float* __restrict__ assignp,
    const int* __restrict__ batch, float* __restrict__ sub,
    float* __restrict__ triv, float* __restrict__ gsum,
    float* __restrict__ counts) {
  int n = blockIdx.x, c = threadIdx.x;
  int g = batch[n];
  float a0 = assignp[(long)n * 2 + 0];
  float a1 = assignp[(long)n * 2 + 1];
  float hv = h[(long)n * HN + c];
  atomicAdd(&sub[(long)g * HN + c], a0 * hv);
  atomicAdd(&triv[(long)g * HN + c], a1 * hv);
  atomicAdd(&gsum[(long)g * HN + c], hv);
  if (c == 0) atomicAdd(&counts[g], 1.0f);
}

__global__ void gin_adj_k(const int* __restrict__ ei, const int* __restrict__ batch,
                          const float* __restrict__ assignp, float* __restrict__ newadj) {
  int e = blockIdx.x * 256 + threadIdx.x;
  if (e >= EE) return;
  int s = ei[e], d = ei[EE + e];
  int gs = batch[s], gd = batch[d];
  if (gs != gd) return;
  float as0 = assignp[(long)s * 2], as1 = assignp[(long)s * 2 + 1];
  float ad0 = assignp[(long)d * 2], ad1 = assignp[(long)d * 2 + 1];
  float* p = newadj + (long)gs * 4;
  atomicAdd(p + 0, as0 * ad0);
  atomicAdd(p + 1, as0 * ad1);
  atomicAdd(p + 2, as1 * ad0);
  atomicAdd(p + 3, as1 * ad1);
}

__global__ void gin_gdiv_k(const float* __restrict__ gsum, const float* __restrict__ counts,
                           float* __restrict__ ge) {
  long i = (long)blockIdx.x * 256 + threadIdx.x;
  if (i >= (long)GG * HN) return;
  int g = (int)(i >> 8);
  ge[i] = gsum[i] / fmaxf(counts[g], 1.0f);
}

__global__ void gin_pen_k(const float* __restrict__ newadj, float* __restrict__ pen) {
  int g = blockIdx.x * 256 + threadIdx.x;
  if (g >= GG) return;
  const float* a = newadj + (long)g * 4;
  float rs0 = fmaxf(fabsf(a[0]) + fabsf(a[1]), 1e-12f);
  float rs1 = fmaxf(fabsf(a[2]) + fabsf(a[3]), 1e-12f);
  float d0 = a[0] / rs0 - 1.0f;
  float d1 = a[3] / rs1 - 1.0f;
  atomicAdd(pen, d0 * d0 + d1 * d1);
}

__global__ void gin_fin_k(const float* __restrict__ pen, float* __restrict__ out) {
  out[0] = pen[0] / (2.0f * (float)GG);
}

// head(e) = log_softmax(relu(e@l1W + l1b) @ l2W + l2b); blockIdx.y: 0=sub, 1=triv
__global__ __launch_bounds__(256) void gin_head_k(
    const float* __restrict__ sub, const float* __restrict__ triv,
    const float* __restrict__ l1W, const float* __restrict__ l1b,
    const float* __restrict__ l2W, const float* __restrict__ l2b,
    float* __restrict__ out) {
  __shared__ float es[256];
  __shared__ float r0[256], r1[256];
  int g = blockIdx.x, sel = blockIdx.y, c = threadIdx.x;
  const float* e = sel ? triv : sub;
  es[c] = e[(long)g * HN + c];
  __syncthreads();
  float acc = l1b[c];
  for (int k = 0; k < HN; ++k) acc += es[k] * l1W[k * HN + c];
  float z = fmaxf(acc, 0.0f);
  r0[c] = z * l2W[c * 2 + 0];
  r1[c] = z * l2W[c * 2 + 1];
  __syncthreads();
  for (int s = 128; s > 0; s >>= 1) {
    if (c < s) { r0[c] += r0[c + s]; r1[c] += r1[c + s]; }
    __syncthreads();
  }
  if (c == 0) {
    float z0 = r0[0] + l2b[0], z1 = r1[0] + l2b[1];
    float m = fmaxf(z0, z1);
    float lse = m + logf(expf(z0 - m) + expf(z1 - m));
    float* dst = out + (long)sel * GG * 2 + (long)g * 2;
    dst[0] = z0 - lse;
    dst[1] = z1 - lse;
  }
}

// ---------------------------------------------------------------------------
extern "C" void kernel_launch(void* const* d_in, const int* in_sizes, int n_in,
                              void* d_out, int out_size, void* d_ws, size_t ws_size,
                              hipStream_t stream) {
  (void)in_sizes; (void)n_in; (void)out_size; (void)ws_size;

  const int*   x        = (const int*)d_in[0];
  const int*   ei       = (const int*)d_in[1];
  const int*   ea       = (const int*)d_in[2];
  const int*   batch    = (const int*)d_in[3];
  const float* atom_emb = (const float*)d_in[5];
  const float* bond_emb = (const float*)d_in[6];
  const float* epsv     = (const float*)d_in[7];
  const float* W1       = (const float*)d_in[8];
  const float* b1       = (const float*)d_in[9];
  const float* gamma    = (const float*)d_in[10];
  const float* beta     = (const float*)d_in[11];
  const float* bmean    = (const float*)d_in[12];
  const float* bvar     = (const float*)d_in[13];
  const float* W2       = (const float*)d_in[14];
  const float* b2       = (const float*)d_in[15];
  const float* c1W      = (const float*)d_in[16];
  const float* c1b      = (const float*)d_in[17];
  const float* c2W      = (const float*)d_in[18];
  const float* c2b      = (const float*)d_in[19];
  const float* l1W      = (const float*)d_in[20];
  const float* l1b      = (const float*)d_in[21];
  const float* l2W      = (const float*)d_in[22];
  const float* l2b      = (const float*)d_in[23];

  // ---- workspace layout ----
  char* w = (char*)d_ws;
  size_t off = 0;
  float*     F0      = (float*)(w + off);     off += (size_t)NN * HN * 4;      // node features
  float*     F1      = (float*)(w + off);     off += (size_t)NN * HN * 4;      // aggregation
  unsigned*  w1p     = (unsigned*)(w + off);  off += (size_t)NL * 32 * 8 * 256 * 4;   // packed W1
  unsigned*  w2p     = (unsigned*)(w + off);  off += (size_t)NL * 16 * 16 * 256 * 4;  // packed W2
  float*     bnscale = (float*)(w + off);     off += (size_t)NL * H2 * 4;
  float*     bnshift = (float*)(w + off);     off += (size_t)NL * H2 * 4;
  float*     assignp = (float*)(w + off);     off += (size_t)NN * 2 * 4;
  float*     triv    = (float*)(w + off);     off += (size_t)GG * HN * 4;      // pool zero-span start
  float*     gsum    = (float*)(w + off);     off += (size_t)GG * HN * 4;
  float*     counts  = (float*)(w + off);     off += (size_t)GG * 4;
  float*     newadj  = (float*)(w + off);     off += (size_t)GG * 4 * 4;
  float*     pen     = (float*)(w + off);     off += 4;

  // ---- output layout (floats, concatenated tuple) ----
  float* out      = (float*)d_out;
  float* out_sub  = out + 2 * (size_t)GG * 2;          // pooled sub features [G,256]
  float* out_ge   = out_sub + (size_t)GG * HN;         // graph_emb [G,256]
  float* out_mask = out_ge + (size_t)GG * HN;          // active_mask [N]
  float* out_pen  = out_mask + (size_t)NN;             // con_penalty [1]

  // prep + zero accumulators (every call: deterministic under graph replay)
  gin_prep_k<<<1280, 256, 0, stream>>>(W1, W2, b1, gamma, beta, bmean, bvar,
                                       w1p, w2p, bnscale, bnshift);
  long poolN = (long)GG * HN * 2 + GG + GG * 4 + 1;    // triv..pen contiguous
  gin_zero_k<<<1024, 256, 0, stream>>>(triv, poolN);
  gin_zero_k<<<1024, 256, 0, stream>>>(out_sub, (long)GG * HN);

  gin_atom_enc<<<NN, 256, 0, stream>>>(x, atom_emb, F0);

  for (int l = 0; l < NL; ++l) {
    gin_zero_k<<<4096, 256, 0, stream>>>(F1, (long)NN * HN);
    gin_edge_msg<<<EE, 256, 0, stream>>>(ei, ea, bond_emb + (size_t)l * 3 * 16 * HN, F0, F1);
    gin_mlp_fused<<<NN / 16, 256, 0, stream>>>(F0, F1, epsv, l, w1p, w2p,
                                               bnscale, bnshift, b2, F0);
  }

  gin_assign_k<<<NN, 256, 0, stream>>>(F0, c1W, c1b, c2W, c2b, assignp, out_mask);
  gin_pool_k<<<NN, 256, 0, stream>>>(F0, assignp, batch, out_sub, triv, gsum, counts);
  gin_adj_k<<<(EE + 255) / 256, 256, 0, stream>>>(ei, batch, assignp, newadj);
  gin_gdiv_k<<<((int)((long)GG * HN + 255) / 256), 256, 0, stream>>>(gsum, counts, out_ge);
  gin_pen_k<<<(GG + 255) / 256, 256, 0, stream>>>(newadj, pen);
  gin_head_k<<<dim3(GG, 2), 256, 0, stream>>>(out_sub, triv, l1W, l1b, l2W, l2b, out);
  gin_fin_k<<<1, 1, 0, stream>>>(pen, out_pen);
}
